// LinkPredictor_69526930588081
// MI455X (gfx1250) — matrix operand order repaired and checked by reading, not compiled, not run on hardware
//
#include <hip/hip_runtime.h>
#include <hip/hip_bf16.h>

typedef __attribute__((ext_vector_type(16))) __bf16 v16bf;
typedef __attribute__((ext_vector_type(8)))  float  v8f;

#define D 256
#define EEDGES 200000
#define KC_N 9              // K padded to 288 (chunks of 32)
#define NC_N 17             // N padded to 272 (chunks of 16)
#define NFRAG (KC_N * NC_N) // 153 B-fragments
#define ASTAGE_BYTES (NFRAG * 1024)

// ---------------- Precompute: t,r -> p,q,c --------------------------------
__global__ void k_vecs(const float* __restrict__ Wb, const float* __restrict__ bd,
                       const float* __restrict__ bs, const float* __restrict__ Ws,
                       const float* __restrict__ Wd, const float* __restrict__ bbil,
                       float* __restrict__ p, float* __restrict__ q, float* __restrict__ c) {
    __shared__ float t[D];
    __shared__ float r[D];
    int i = threadIdx.x;
    float ti = 0.f, ri = 0.f;
    for (int j = 0; j < D; ++j) {
        ti += Wb[i * D + j] * bd[j];   // t[i] = sum_j Wb[i,j] bd[j]
        ri += bs[j] * Wb[j * D + i];   // r[i] = sum_j bs[j] Wb[j,i]
    }
    t[i] = ti; r[i] = ri;
    __syncthreads();
    float pi = 0.f, qi = 0.f;
    for (int k = 0; k < D; ++k) {
        pi += Ws[k * D + i] * t[k];    // p[a] = sum_i Ws[i,a] t[i]
        qi += r[k] * Wd[k * D + i];    // q[b] = sum_j r[j] Wd[j,b]
    }
    p[i] = pi; q[i] = qi;
    if (i == 0) {
        float cc = 0.f;
        for (int j = 0; j < D; ++j) cc += r[j] * bd[j];
        *c = cc + bbil[0];
    }
}

// M = Wb @ Wd  (M[i,b] = sum_j Wb[i,j] Wd[j,b])
__global__ void k_M(const float* __restrict__ Wb, const float* __restrict__ Wd,
                    float* __restrict__ M) {
    __shared__ float row[D];
    int i = blockIdx.x, b = threadIdx.x;
    row[b] = Wb[i * D + b];
    __syncthreads();
    float acc = 0.f;
    for (int j = 0; j < D; ++j) acc += row[j] * Wd[j * D + b];
    M[i * D + b] = acc;
}

// A = Ws^T @ M  (A[a,b] = sum_i Ws[i,a] M[i,b])
__global__ void k_A(const float* __restrict__ Ws, const float* __restrict__ M,
                    float* __restrict__ A) {
    __shared__ float col[D];
    int a = blockIdx.x, b = threadIdx.x;
    col[b] = Ws[b * D + a];
    __syncthreads();
    float acc = 0.f;
    for (int i = 0; i < D; ++i) acc += col[i] * M[i * D + b];
    A[a * D + b] = acc;
}

// Stage Atilde (288x272; row 256 = q/c, col 256 = p) into bf16 B-fragments.
// B-fragment (32x16, 16-bit): lane<16 holds K=0..15 of column N=lane,
// lane>=16 holds K=16..31 of column N=lane-16.
// LDS layout per fragment (1024B): half0 (elements j=0..7 of each lane) at
// f*1024 + lane*16; half1 (j=8..15) at f*1024 + 512 + lane*16 -> 16B lane
// stride => conflict-free ds_load_b128.
__global__ void k_stage(const float* __restrict__ A, const float* __restrict__ p,
                        const float* __restrict__ q, const float* __restrict__ c,
                        __bf16* __restrict__ Ast) {
    int f = blockIdx.x;               // 0..NFRAG-1
    int kc = f / NC_N, nc = f % NC_N;
    int lane = threadIdx.x;           // 32 lanes
    int col = nc * 16 + (lane & 15);
    int kbase = kc * 32 + ((lane < 16) ? 0 : 16);
    for (int j = 0; j < 16; ++j) {
        int k = kbase + j;
        float v = 0.f;
        if (k < D && col < D)        v = A[k * D + col];
        else if (k == D && col < D)  v = q[col];
        else if (col == D && k < D)  v = p[k];
        else if (k == D && col == D) v = *c;
        int idx = (j < 8) ? (f * 512 + lane * 8 + j)
                          : (f * 512 + 256 + lane * 8 + (j - 8));
        Ast[idx] = (__bf16)v;
    }
}

union FragU { uint4 u[2]; v16bf v; };

// ---------------- Main: per-16-edge-tile bilinear scores via WMMA ----------
__global__ __launch_bounds__(256)
void k_score(const float* __restrict__ x,
             const long long* __restrict__ epos,
             const long long* __restrict__ eneg,
             const __bf16* __restrict__ Ast,
             float* __restrict__ out, int nblocks) {
    extern __shared__ unsigned char ldsRaw[];
    uint4* ldsA = (uint4*)ldsRaw;
    const uint4* Ag = (const uint4*)Ast;

    int tid = threadIdx.x;
    for (int i = tid; i < ASTAGE_BYTES / 16; i += 256) ldsA[i] = Ag[i];
    __syncthreads();

    int set = blockIdx.y;
    const long long* eidx = set ? eneg : epos;
    const long long* srcI = eidx;             // edge_index[0]
    const long long* dstI = eidx + EEDGES;    // edge_index[1]
    float* oset = out + 1 + set * EEDGES;

    int lane = tid & 31;
    int wave = tid >> 5;
    int e16  = lane & 15;
    bool hi  = lane >= 16;

    const int ntiles = EEDGES / 16;
    for (int t = blockIdx.x * 8 + wave; t < ntiles; t += nblocks * 8) {
        // Opaque zero: makes LDS fragment addresses loop-variant so LICM
        // cannot hoist (and then spill) the B-fragment loads out of this loop.
        int ldsOfs = 0;
        asm volatile("" : "+v"(ldsOfs));

        int base = t * 16;
        int srow = (int)srcI[base + e16];
        const float* urow = x + (long long)srow * D;

        // Build all 9 A-fragments (16x32 bf16) once; keep resident (~72 VGPRs).
        // lane<16 -> K kc*32+{0..15} of row e16; lane>=16 -> {16..31}.
        v16bf uf[KC_N];
        #pragma unroll
        for (int kc = 0; kc < KC_N; ++kc) {
            if (kc < 8) {
                int kb = kc * 32 + (hi ? 16 : 0);
                float4 a0 = *(const float4*)(urow + kb);
                float4 a1 = *(const float4*)(urow + kb + 4);
                float4 b0 = *(const float4*)(urow + kb + 8);
                float4 b1 = *(const float4*)(urow + kb + 12);
                uf[kc][0]  = (__bf16)a0.x; uf[kc][1]  = (__bf16)a0.y;
                uf[kc][2]  = (__bf16)a0.z; uf[kc][3]  = (__bf16)a0.w;
                uf[kc][4]  = (__bf16)a1.x; uf[kc][5]  = (__bf16)a1.y;
                uf[kc][6]  = (__bf16)a1.z; uf[kc][7]  = (__bf16)a1.w;
                uf[kc][8]  = (__bf16)b0.x; uf[kc][9]  = (__bf16)b0.y;
                uf[kc][10] = (__bf16)b0.z; uf[kc][11] = (__bf16)b0.w;
                uf[kc][12] = (__bf16)b1.x; uf[kc][13] = (__bf16)b1.y;
                uf[kc][14] = (__bf16)b1.z; uf[kc][15] = (__bf16)b1.w;
            } else {
                // padded K chunk: only U[e,256] = 1 (lane<16, element 0)
                #pragma unroll
                for (int j = 0; j < 16; ++j) uf[kc][j] = (__bf16)0.f;
                if (!hi) uf[kc][0] = (__bf16)1.0f;
            }
        }

        int dIdx[8];
        #pragma unroll
        for (int r = 0; r < 8; ++r)
            dIdx[r] = (int)dstI[base + (hi ? 8 : 0) + r];

        float ps[8];
        #pragma unroll
        for (int r = 0; r < 8; ++r) ps[r] = 0.f;

        // N-chunks in pairs: two independent WMMA chains, 2 live accumulators.
        #pragma clang loop unroll(disable)
        for (int ncp = 0; ncp < 8; ++ncp) {
            v8f acc0 = {}, acc1 = {};
            #pragma unroll
            for (int kc = 0; kc < KC_N; ++kc) {
                const uint4* fp = ldsA + ldsOfs + (kc * NC_N + 2 * ncp) * 64 + lane;
                FragU b0, b1;
                b0.u[0] = fp[0];   b0.u[1] = fp[32];   // fragment 2*ncp
                b1.u[0] = fp[64];  b1.u[1] = fp[96];   // fragment 2*ncp+1
                acc0 = __builtin_amdgcn_wmma_f32_16x16x32_bf16(
                    false, uf[kc], false, b0.v, (short)0, acc0, false, false);
                acc1 = __builtin_amdgcn_wmma_f32_16x16x32_bf16(
                    false, uf[kc], false, b1.v, (short)0, acc1, false, false);
            }
            // epilogue for nc = 2*ncp, 2*ncp+1 (V side in f32, coalesced 64B)
            int bc = (2 * ncp) * 16 + e16;
            #pragma unroll
            for (int r = 0; r < 8; ++r) {
                const float* vrow = x + (long long)dIdx[r] * D;
                ps[r] += acc0[r] * vrow[bc] + acc1[r] * vrow[bc + 16];
            }
        }

        // Tail N-chunk (nc = 16): Vtilde[e,256] = 1, rest 0
        {
            v8f accT = {};
            #pragma unroll
            for (int kc = 0; kc < KC_N; ++kc) {
                const uint4* fp = ldsA + ldsOfs + (kc * NC_N + 16) * 64 + lane;
                FragU bt;
                bt.u[0] = fp[0]; bt.u[1] = fp[32];
                accT = __builtin_amdgcn_wmma_f32_16x16x32_bf16(
                    false, uf[kc], false, bt.v, (short)0, accT, false, false);
            }
            float vT = (e16 == 0) ? 1.0f : 0.0f;
            #pragma unroll
            for (int r = 0; r < 8; ++r) ps[r] += accT[r] * vT;
        }

        // reduce across the 16 lanes of each half-wave (edges 0..7 / 8..15)
        #pragma unroll
        for (int r = 0; r < 8; ++r) {
            float s = ps[r];
            s += __shfl_xor(s, 1, 32);
            s += __shfl_xor(s, 2, 32);
            s += __shfl_xor(s, 4, 32);
            s += __shfl_xor(s, 8, 32);
            ps[r] = s;
        }
        if (e16 == 0) {                 // lanes 0 and 16 write 8 edges each
            int ebase = base + (hi ? 8 : 0);
            #pragma unroll
            for (int r = 0; r < 8; ++r) {
                float sc = 1.0f / (1.0f + __expf(-ps[r]));
                oset[ebase + r] = sc;
            }
        }
    }
}

// ---------------- Loss: deterministic two-stage reduction ------------------
__global__ void k_loss_partial(const float* __restrict__ out, float* __restrict__ partials) {
    __shared__ float sm[256];
    const float* pos = out + 1;
    const float* neg = out + 1 + EEDGES;
    int chunk = (EEDGES + gridDim.x - 1) / gridDim.x;
    int s0 = blockIdx.x * chunk;
    int s1 = s0 + chunk; if (s1 > EEDGES) s1 = EEDGES;
    float a = 0.f;
    for (int i = s0 + threadIdx.x; i < s1; i += blockDim.x)
        a += fmaxf(1.0f - pos[i] + neg[i], 0.0f);
    sm[threadIdx.x] = a;
    __syncthreads();
    for (int off = 128; off > 0; off >>= 1) {
        if ((int)threadIdx.x < off) sm[threadIdx.x] += sm[threadIdx.x + off];
        __syncthreads();
    }
    if (threadIdx.x == 0) partials[blockIdx.x] = sm[0];
}

__global__ void k_loss_final(const float* __restrict__ partials, float* __restrict__ out) {
    __shared__ float sm[256];
    sm[threadIdx.x] = partials[threadIdx.x];
    __syncthreads();
    for (int off = 128; off > 0; off >>= 1) {
        if ((int)threadIdx.x < off) sm[threadIdx.x] += sm[threadIdx.x + off];
        __syncthreads();
    }
    if (threadIdx.x == 0) out[0] = sm[0] / (float)EEDGES;
}

// ---------------- Launch ---------------------------------------------------
extern "C" void kernel_launch(void* const* d_in, const int* in_sizes, int n_in,
                              void* d_out, int out_size, void* d_ws, size_t ws_size,
                              hipStream_t stream) {
    (void)in_sizes; (void)n_in; (void)out_size; (void)ws_size;
    const float*     x    = (const float*)d_in[0];
    const long long* pe   = (const long long*)d_in[1];  // int64 [2,E]
    const long long* ne   = (const long long*)d_in[2];
    const float*     Ws   = (const float*)d_in[3];
    const float*     bs   = (const float*)d_in[4];
    const float*     Wd   = (const float*)d_in[5];
    const float*     bd   = (const float*)d_in[6];
    const float*     Wb   = (const float*)d_in[7];      // [1,D,D]
    const float*     bbil = (const float*)d_in[8];
    float* out = (float*)d_out;

    char* ws = (char*)d_ws;
    float*  M        = (float*)(ws + 0);
    float*  A        = (float*)(ws + 262144);
    float*  p        = (float*)(ws + 524288);
    float*  q        = (float*)(ws + 525312);
    float*  c        = (float*)(ws + 526336);
    __bf16* Ast      = (__bf16*)(ws + 528384);
    float*  partials = (float*)(ws + 528384 + ASTAGE_BYTES);

    k_vecs<<<1, 256, 0, stream>>>(Wb, bd, bs, Ws, Wd, bbil, p, q, c);
    k_M<<<256, 256, 0, stream>>>(Wb, Wd, M);
    k_A<<<256, 256, 0, stream>>>(Ws, M, A);
    k_stage<<<NFRAG, 32, 0, stream>>>(A, p, q, c, Ast);

    const int nblocks = 256;
    dim3 grid(nblocks, 2);
    k_score<<<grid, 256, ASTAGE_BYTES, stream>>>(x, pe, ne, Ast, out, nblocks);

    k_loss_partial<<<256, 256, 0, stream>>>(out, partials);
    k_loss_final<<<1, 256, 0, stream>>>(partials, out);
}
